// MyLSTM_30434138259937
// MI455X (gfx1250) — compile-verified, matrix operand order checked by reading, not run on hardware
//
#include <hip/hip_runtime.h>
#include <hip/hip_bf16.h>
#include <stdint.h>

// ---------------------------------------------------------------------------
// Problem constants (match the JAX reference)
// ---------------------------------------------------------------------------
#define BB    128        // batch
#define TT    512        // time steps
#define DD    128        // input feature dim
#define HH    256        // hidden dim
#define GG    1024       // 4*H gates
#define MT0   (BB*TT)    // rows of the big input-projection GEMMs
#define SLAB  (GG*16)    // one (t, 16-batch-chunk) xg slab: 16384 f32 = 64 KB

typedef __attribute__((ext_vector_type(16))) __bf16       v16bf;
typedef __attribute__((ext_vector_type(8)))  __bf16       v8bf;
typedef __attribute__((ext_vector_type(8)))  float        v8f;
typedef __attribute__((ext_vector_type(4)))  unsigned int v4u;
typedef __attribute__((ext_vector_type(8)))  int          v8i;
typedef __attribute__((ext_vector_type(4)))  int          v4i;

// ---------------------------------------------------------------------------
// Helpers
// ---------------------------------------------------------------------------
__device__ __forceinline__ __bf16 f2bf(float f) {
    union { float f; uint32_t u; } x; x.f = f;
    uint32_t r = x.u + 0x7FFFu + ((x.u >> 16) & 1u);   // round-to-nearest-even
    union { uint16_t u; __bf16 b; } y; y.u = (uint16_t)(r >> 16);
    return y.b;
}
__device__ __forceinline__ float sigf(float x)  { return 1.0f / (1.0f + __expf(-x)); }
__device__ __forceinline__ float tanhf_(float x){ float e = __expf(2.0f * x); return (e - 1.0f) / (e + 1.0f); }

__device__ __forceinline__ v16bf ldfrag(const __bf16* p) {
    const v8bf lo = *(const v8bf*)(p);
    const v8bf hi = *(const v8bf*)(p + 16);
    v16bf a;
#pragma unroll
    for (int i = 0; i < 8; ++i) { a[i] = lo[i]; a[i + 8] = hi[i]; }
    return a;
}

// ---------------------------------------------------------------------------
// TDM: async-DMA one contiguous 64 KB xg slab (16384 f32) global -> LDS.
// D# per CDNA5 ISA: count=1, data_size=4B (code 2),
// 1-D tensor/tile of 16384 elements, descriptor type=2 ("image").
// This toolchain exposes the 6-arg builtin (extra unused int32x8 group).
// ---------------------------------------------------------------------------
__device__ __forceinline__ void tdm_load_slab(const float* gsrc, uint32_t lds_off) {
    const uint64_t ga = (uint64_t)(uintptr_t)gsrc;
    v4u g0;
    g0.x = 1u;                                                   // count=1, user mode
    g0.y = lds_off;                                              // lds_addr
    g0.z = (uint32_t)ga;                                         // global_addr[31:0]
    g0.w = (uint32_t)((ga >> 32) & 0x01FFFFFFu) | (2u << 30);    // addr[56:32] | type=2
    v8i g1;
    g1[0] = 0x00020000;               // workgroup_mask=0, data_size=2 (4 bytes)
    g1[1] = (int)(16384u << 16);      // tensor_dim0[15:0] -> bits 63:48
    g1[2] = (int)(1u << 16);          // tensor_dim0 hi=0 ; tensor_dim1=1
    g1[3] = (int)(16384u << 16);      // tile_dim0 = 16384 -> bits 127:112
    g1[4] = 1;                        // tile_dim1 = 1, tile_dim2 = 0
    g1[5] = 16384;                    // tensor_dim0_stride low 32
    g1[6] = (int)(16384u << 16);      // stride hi=0 ; tensor_dim1_stride lo16
    g1[7] = 0;
    const v4i z4 = {0, 0, 0, 0};
    const v8i z8 = {0, 0, 0, 0, 0, 0, 0, 0};
    __builtin_amdgcn_tensor_load_to_lds(g0, g1, z4, z4, z8, 0);
}

// ---------------------------------------------------------------------------
// fp32 -> bf16 conversion (grid-stride)
// ---------------------------------------------------------------------------
__global__ void cvt_bf16_kernel(const float* __restrict__ src,
                                __bf16* __restrict__ dst, int n) {
    for (int i = blockIdx.x * blockDim.x + threadIdx.x; i < n;
         i += gridDim.x * blockDim.x)
        dst[i] = f2bf(src[i]);
}

// ---------------------------------------------------------------------------
// Input-projection GEMM into fragment-layout xg:
//   logical row r = t*BB + b ; out[(r/16)*GG + g][m] = A[r].Wih[g] + b1[g]+b2[g]
//   A: bf16, mode 0: X is [B,T,D] -> row gather (b*TT + t); mode 1: identity.
//   Each wave computes a 16x64 tile strip: one A-frag feeds 4 WMMAs per K-step.
//   Output element (tile rt, gate g, row m) at  xgf[(rt*GG + g)*16 + m]
//   -> per-lane stores are 8 contiguous floats (2x global_store_b128).
// ---------------------------------------------------------------------------
__global__ __launch_bounds__(256)
void xg_gemm_kernel(const __bf16* __restrict__ A,
                    const __bf16* __restrict__ W,
                    const float* __restrict__ bias1,
                    const float* __restrict__ bias2,
                    float* __restrict__ out,
                    int M, int K, int mode) {
    const int lane = threadIdx.x & 31;
    const int wave = threadIdx.x >> 5;
    const int wid  = blockIdx.x * 8 + wave;
    const int nGrp = GG / 64;                  // 16 groups of 4 N-tiles
    const int mt   = wid / nGrp;
    const int g0   = (wid % nGrp) * 64;
    if (mt * 16 >= M) return;

    const int r0   = mt * 16;
    const int n    = lane & 15;                // column / B-row within tile
    const int m    = lane & 15;                // A row within tile
    const int c0   = (lane < 16) ? 0 : 8;      // K-chunk select (ISA layout)
    const int hi8  = (lane < 16) ? 0 : 8;      // C-fragment row offset

    // A row pointer (per-lane gather handles the layer-0 [B,T,D] layout)
    const __bf16* arow;
    if (mode == 0) {
        const int t = r0 / BB;
        const int b = (r0 & (BB - 1)) + m;
        arow = A + ((size_t)b * TT + t) * K;
    } else {
        arow = A + (size_t)(r0 + m) * K;
    }

    v8f acc[4];
#pragma unroll
    for (int tt = 0; tt < 4; ++tt) {
        const float bsum = bias1[g0 + tt * 16 + n] + bias2[g0 + tt * 16 + n];
#pragma unroll
        for (int e = 0; e < 8; ++e) acc[tt][e] = bsum;
    }

    for (int k0 = 0; k0 < K; k0 += 32) {
        const v16bf a = ldfrag(arow + k0 + c0);
#pragma unroll
        for (int tt = 0; tt < 4; ++tt) {
            const v16bf b = ldfrag(W + (size_t)(g0 + tt * 16 + n) * K + k0 + c0);
            acc[tt] = __builtin_amdgcn_wmma_f32_16x16x32_bf16(
                          false, a, false, b, (short)0, acc[tt], false, false);
        }
    }

    const int rt = r0 >> 4;
#pragma unroll
    for (int tt = 0; tt < 4; ++tt) {
        float* op = out + ((size_t)rt * GG + (g0 + tt * 16 + n)) * 16 + hi8;
#pragma unroll
        for (int e = 0; e < 8; ++e) op[e] = acc[tt][e];   // 2x b128 stores
    }
}

// ---------------------------------------------------------------------------
// LSTM scan: 8 blocks x 16 batch rows, persistent over all T steps.
//   xgf: fp32 fragment layout [T*8 tiles][GG][16] (includes both biases)
//   W:   bf16 [GG, HH] recurrent weights (L2 resident)
//   hseq_bf: bf16 [T, B, HH] full h sequence (layer 0) or nullptr
//   hlast:   fp32 [B, HH] final h (layer 1) or nullptr
// 512 threads = 16 waves (4 per SIMD32) -> 16 independent load/WMMA streams
// per step to hide L2 latency on the Whh B-fragments. Each wave owns a
// 16-wide H slice (one tile per gate) so the c update is register-local.
// TDM double-buffers the per-step 64 KB xg slab into LDS while WMMAs run.
// ---------------------------------------------------------------------------
__global__ __launch_bounds__(512)
void lstm_scan_kernel(const float* __restrict__ xgf,
                      const __bf16* __restrict__ W,
                      __bf16* __restrict__ hseq_bf,
                      float* __restrict__ hlast) {
    __shared__ float  xls[2][SLAB];            // 2 x 64 KB xg slabs (TDM dest)
    __shared__ __bf16 hbf[2][16][HH + 8];      // double-buffered h, padded rows

    const int blk  = blockIdx.x;
    const int b0   = blk * 16;
    const int lane = threadIdx.x & 31;
    const int wave = threadIdx.x >> 5;
    const int k0   = wave * 16;                // this wave's 16-wide H-slice
    const int n    = lane & 15;
    const int c0   = (lane < 16) ? 0 : 8;
    const int hi8  = (lane < 16) ? 0 : 8;
    const int mrow = lane & 15;

    // zero-init h buffers
    for (int i = threadIdx.x; i < 2 * 16 * (HH + 8); i += blockDim.x)
        ((__bf16*)hbf)[i] = f2bf(0.0f);

    // TDM-prefetch slab for t=0 into buffer 0
    if (wave == 0) {
        tdm_load_slab(xgf + (size_t)blk * SLAB, (uint32_t)(uintptr_t)&xls[0][0]);
        __builtin_amdgcn_s_wait_tensorcnt(0);
    }
    __syncthreads();

    float creg[8];
#pragma unroll
    for (int e = 0; e < 8; ++e) creg[e] = 0.0f;

    for (int t = 0; t < TT; ++t) {
        const int buf = t & 1;

        // kick off async DMA of next step's slab into the other buffer
        if (wave == 0 && t + 1 < TT)
            tdm_load_slab(xgf + (size_t)((t + 1) * 8 + blk) * SLAB,
                          (uint32_t)(uintptr_t)&xls[buf ^ 1][0]);

        // init accumulators from the LDS xg slab (2x ds_load_b128 per tile)
        v8f acc[4];
#pragma unroll
        for (int q = 0; q < 4; ++q) {
            const int col = q * HH + k0 + n;
            const float* xp = &xls[buf][(col << 4) + hi8];
#pragma unroll
            for (int e = 0; e < 8; ++e) acc[q][e] = xp[e];
        }

        // recurrent GEMM: h[16,HH] x Whh^T slice -> 4 gate tiles x 8 K-steps
#pragma unroll
        for (int kk = 0; kk < HH; kk += 32) {
            const v16bf a = ldfrag(&hbf[buf][mrow][kk + c0]);
#pragma unroll
            for (int q = 0; q < 4; ++q) {
                const v16bf b = ldfrag(
                    W + (size_t)(q * HH + k0 + n) * HH + kk + c0);
                acc[q] = __builtin_amdgcn_wmma_f32_16x16x32_bf16(
                             false, a, false, b, (short)0, acc[q], false, false);
            }
        }

        // gate nonlinearities + state update (cross-gate combine in registers)
#pragma unroll
        for (int e = 0; e < 8; ++e) {
            const float iv = sigf(acc[0][e]);
            const float fv = sigf(acc[1][e]);
            const float gv = tanhf_(acc[2][e]);
            const float ov = sigf(acc[3][e]);
            const float c  = fv * creg[e] + iv * gv;
            creg[e] = c;
            const float h  = ov * tanhf_(c);
            const int   m  = e + hi8;
            const int  col = k0 + n;
            const __bf16 hb = f2bf(h);
            hbf[buf ^ 1][m][col] = hb;
            if (hseq_bf)
                hseq_bf[((size_t)t * BB + b0 + m) * HH + col] = hb;
            if (hlast && t == TT - 1)
                hlast[(size_t)(b0 + m) * HH + col] = h;
        }

        // ensure next slab landed, then release everyone into step t+1
        if (wave == 0) __builtin_amdgcn_s_wait_tensorcnt(0);
        __syncthreads();
    }
}

// ---------------------------------------------------------------------------
// Final FC: out[b] = h2last[b,:] . Wfc + bfc     (O = 1)
// ---------------------------------------------------------------------------
__global__ void fc_kernel(const float* __restrict__ h,
                          const float* __restrict__ Wfc,
                          const float* __restrict__ bfc,
                          float* __restrict__ out) {
    int b = blockIdx.x * blockDim.x + threadIdx.x;
    if (b < BB) {
        float s = bfc[0];
#pragma unroll 4
        for (int k = 0; k < HH; ++k) s += h[(size_t)b * HH + k] * Wfc[k];
        out[b] = s;
    }
}

// ---------------------------------------------------------------------------
// Host launcher
// ---------------------------------------------------------------------------
extern "C" void kernel_launch(void* const* d_in, const int* in_sizes, int n_in,
                              void* d_out, int out_size, void* d_ws, size_t ws_size,
                              hipStream_t stream) {
    const float* X    = (const float*)d_in[0];
    const float* Wih0 = (const float*)d_in[1];
    const float* Whh0 = (const float*)d_in[2];
    const float* bih0 = (const float*)d_in[3];
    const float* bhh0 = (const float*)d_in[4];
    const float* Wih1 = (const float*)d_in[5];
    const float* Whh1 = (const float*)d_in[6];
    const float* bih1 = (const float*)d_in[7];
    const float* bhh1 = (const float*)d_in[8];
    const float* Wfc  = (const float*)d_in[9];
    const float* bfc  = (const float*)d_in[10];
    float* out = (float*)d_out;

    // workspace carve-up
    char* p = (char*)d_ws;
    float*  xgf    = (float*)p;   p += (size_t)TT * BB * GG * 4;   // 268 MB, reused by both layers
    __bf16* h1bf   = (__bf16*)p;  p += (size_t)TT * BB * HH * 2;   // layer-0 h sequence, bf16 [T,B,H]
    float*  h2last = (float*)p;   p += (size_t)BB * HH * 4;
    __bf16* Xbf    = (__bf16*)p;  p += (size_t)BB * TT * DD * 2;
    __bf16* Wih0bf = (__bf16*)p;  p += (size_t)GG * DD * 2;
    __bf16* Whh0bf = (__bf16*)p;  p += (size_t)GG * HH * 2;
    __bf16* Wih1bf = (__bf16*)p;  p += (size_t)GG * HH * 2;
    __bf16* Whh1bf = (__bf16*)p;  p += (size_t)GG * HH * 2;

    auto cvt = [&](const float* s, __bf16* d, int n) {
        int blocks = (n + 255) / 256; if (blocks > 4096) blocks = 4096;
        cvt_bf16_kernel<<<blocks, 256, 0, stream>>>(s, d, n);
    };

    // 1) bf16 operand prep
    cvt(X,    Xbf,    BB * TT * DD);
    cvt(Wih0, Wih0bf, GG * DD);
    cvt(Whh0, Whh0bf, GG * HH);
    cvt(Wih1, Wih1bf, GG * HH);
    cvt(Whh1, Whh1bf, GG * HH);

    const int gemmGrid = ((MT0 / 16) * (GG / 64)) / 8;   // 65536 waves / 8 per block

    // 2) layer-0 input projection -> fragment-layout xg (+ both biases)
    xg_gemm_kernel<<<gemmGrid, 256, 0, stream>>>(Xbf, Wih0bf, bih0, bhh0,
                                                 xgf, MT0, DD, /*mode=*/0);
    // 3) layer-0 scan (writes bf16 h sequence directly)
    lstm_scan_kernel<<<BB / 16, 512, 0, stream>>>(xgf, Whh0bf, h1bf, nullptr);

    // 4) layer-1 input projection (h1bf is [T,B,H] -> identity row map)
    xg_gemm_kernel<<<gemmGrid, 256, 0, stream>>>(h1bf, Wih1bf, bih1, bhh1,
                                                 xgf, MT0, HH, /*mode=*/1);
    // 5) layer-1 scan, keep only last h
    lstm_scan_kernel<<<BB / 16, 512, 0, stream>>>(xgf, Whh1bf, nullptr, h2last);

    // 6) final FC
    fc_kernel<<<1, BB, 0, stream>>>(h2last, Wfc, bfc, out);
}
